// AttentionBlock_58652073394449
// MI455X (gfx1250) — matrix operand order repaired and checked by reading, not compiled, not run on hardware
//
#include <hip/hip_runtime.h>
#include <hip/hip_bf16.h>

typedef __attribute__((ext_vector_type(16))) __bf16 v16bf;
typedef __attribute__((ext_vector_type(8)))  float  v8f;
typedef __attribute__((ext_vector_type(4)))  unsigned int v4u;
typedef __attribute__((ext_vector_type(4)))  int v4i;

// address-space qualified int4 for the async-to-LDS builtin
typedef __attribute__((address_space(1))) v4i av4i_g;   // global
typedef __attribute__((address_space(3))) v4i av4i_l;   // LDS

union Frag16 { v16bf v; v4u q[2]; };

#define WMMA_BF16(A, B, C) \
  __builtin_amdgcn_wmma_f32_16x16x32_bf16(false, (A), false, (B), (short)0, (C), false, false)

// ---------- helpers ----------
__device__ __forceinline__ float redmax16(float x) {
  x = fmaxf(x, __shfl_xor(x, 1, 32));
  x = fmaxf(x, __shfl_xor(x, 2, 32));
  x = fmaxf(x, __shfl_xor(x, 4, 32));
  x = fmaxf(x, __shfl_xor(x, 8, 32));
  return x;
}
__device__ __forceinline__ float redsum16(float x) {
  x += __shfl_xor(x, 1, 32);
  x += __shfl_xor(x, 2, 32);
  x += __shfl_xor(x, 4, 32);
  x += __shfl_xor(x, 8, 32);
  return x;
}

// A fragment: 16x32 bf16, rows striped per ISA layout (lane row = lane&15,
// K halves split by lane>>4). rowp points at the row's K-chunk start.
__device__ __forceinline__ v16bf load_a(const char* rowp, int half) {
  Frag16 f;
  f.q[0] = *(const v4u*)(rowp + 16 * half);
  f.q[1] = *(const v4u*)(rowp + 32 + 16 * half);
  return f.v;
}
// B fragment from B^T stored row-major (row = output col n, 32 contiguous K elems).
__device__ __forceinline__ v16bf load_b(const char* rowp, int half) {
  Frag16 f;
  f.q[0] = *(const v4u*)(rowp + 16 * half);
  f.q[1] = *(const v4u*)(rowp + 32 + 16 * half);
  return f.v;
}

// ---------- kernel 0: out[:, :1024] = x ----------
__global__ void copy_x_kernel(const float* __restrict__ x, float* __restrict__ out) {
  size_t idx = (size_t)blockIdx.x * blockDim.x + threadIdx.x;  // over 4096*256 float4
  int t  = (int)(idx >> 8);
  int c4 = (int)(idx & 255);
  ((float4*)out)[(size_t)t * 512 + c4] = ((const float4*)x)[idx];
}

// ---------- kernel 1: projections Q,K row-major bf16; V transposed bf16 ----------
__global__ void proj_kernel(const float* __restrict__ x,
                            const float* __restrict__ Wk, const float* __restrict__ bk,
                            const float* __restrict__ Wq, const float* __restrict__ bq,
                            const float* __restrict__ Wv, const float* __restrict__ bv,
                            __bf16* __restrict__ Kb, __bf16* __restrict__ Qb,
                            __bf16* __restrict__ Vt) {
  const int mat = blockIdx.z;
  const float* W    = (mat == 0) ? Wk : (mat == 1) ? Wq : Wv;
  const float* bias = (mat == 0) ? bk : (mat == 1) ? bq : bv;
  const int n0 = blockIdx.x * 64;
  const int t0 = blockIdx.y * 64;
  const int tid = threadIdx.x;
  const int wave = tid >> 5, lane = tid & 31;
  const int half = lane >> 4, nidx = lane & 15;

  __shared__ __bf16 xs[64 * 32];
  __shared__ __bf16 wsh[64 * 32];

  v8f acc[4] = {};

  const int lrow = tid >> 1;
  const int lcol = (tid & 1) * 16;

  for (int k0 = 0; k0 < 1024; k0 += 32) {
    __syncthreads();
    {
      const float4* sx = (const float4*)(x + (size_t)(t0 + lrow) * 1024 + k0 + lcol);
      const float4* sw = (const float4*)(W + (size_t)(n0 + lrow) * 1024 + k0 + lcol);
#pragma unroll
      for (int q = 0; q < 4; ++q) {
        float4 f = sx[q];
        int o = lrow * 32 + lcol + q * 4;
        xs[o + 0] = (__bf16)f.x; xs[o + 1] = (__bf16)f.y;
        xs[o + 2] = (__bf16)f.z; xs[o + 3] = (__bf16)f.w;
        float4 g = sw[q];
        wsh[o + 0] = (__bf16)g.x; wsh[o + 1] = (__bf16)g.y;
        wsh[o + 2] = (__bf16)g.z; wsh[o + 3] = (__bf16)g.w;
      }
    }
    __syncthreads();
    v16bf a  = load_a((const char*)(xs + (wave * 16 + nidx) * 32), half);
    v16bf b0 = load_b((const char*)(wsh + ( 0 + nidx) * 32), half);
    v16bf b1 = load_b((const char*)(wsh + (16 + nidx) * 32), half);
    v16bf b2 = load_b((const char*)(wsh + (32 + nidx) * 32), half);
    v16bf b3 = load_b((const char*)(wsh + (48 + nidx) * 32), half);
    acc[0] = WMMA_BF16(a, b0, acc[0]);
    acc[1] = WMMA_BF16(a, b1, acc[1]);
    acc[2] = WMMA_BF16(a, b2, acc[2]);
    acc[3] = WMMA_BF16(a, b3, acc[3]);
  }

  // branch-free store: K/Q row-major (rs=1024, cs=1); V transposed (rs=1, cs=4096)
  __bf16* dst = (mat == 0) ? Kb : (mat == 1) ? Qb : Vt;
  const size_t rs = (mat == 2) ? (size_t)1 : (size_t)1024;
  const size_t cs = (mat == 2) ? (size_t)4096 : (size_t)1;
#pragma unroll
  for (int nt = 0; nt < 4; ++nt) {
    int n = n0 + nt * 16 + nidx;
    float bb = bias[n];
#pragma unroll
    for (int v = 0; v < 8; ++v) {
      int t = t0 + wave * 16 + v + 8 * half;
      dst[(size_t)t * rs + (size_t)n * cs] = (__bf16)(acc[nt][v] + bb);
    }
  }
}

// score tile: 16x16 scores for (row tile rt cached in LDS qtile) x (col tile jt)
__device__ __forceinline__ v8f score_tile(const __bf16* qtile, const __bf16* __restrict__ Kb,
                                          int jt, int half, int nidx) {
  v8f acc = {};
  const char* arow = (const char*)(qtile + nidx * 1024);
  const char* brow = (const char*)(Kb + (size_t)(jt * 16 + nidx) * 1024);
#pragma unroll 4
  for (int kt = 0; kt < 32; ++kt) {
    v16bf a = load_a(arow + kt * 64, half);
    v16bf b = load_b(brow + kt * 64, half);
    acc = WMMA_BF16(a, b, acc);
  }
  return acc;
}

// ---------- kernel 2: masked softmax(Q K^T / sqrt(dk)) -> Pb (bf16) ----------
__global__ void attn_softmax_kernel(const __bf16* __restrict__ Qb,
                                    const __bf16* __restrict__ Kb,
                                    __bf16* __restrict__ Pb) {
  const int rt = blockIdx.x;                 // 16-row query tile, 0..255
  const int tid = threadIdx.x;
  const int wave = tid >> 5, lane = tid & 31;
  const int half = lane >> 4, nidx = lane & 15;

  __shared__ __bf16 qtile[16 * 1024];        // 32KB
  __shared__ float mws[4][16];
  __shared__ float lws[4][16];

  // cache Q row tile in LDS; use CDNA5 async memory->LDS path when available
  {
    const v4u* src = (const v4u*)(Qb + (size_t)rt * 16 * 1024);
    v4u* dst = (v4u*)qtile;
#if defined(__gfx1250__) && __has_builtin(__builtin_amdgcn_global_load_async_to_lds_b128)
    for (int i = tid; i < 2048; i += 128) {
      __builtin_amdgcn_global_load_async_to_lds_b128(
          (av4i_g*)(src + i), (av4i_l*)(dst + i), /*offset=*/0, /*cpol=*/0);
    }
#if __has_builtin(__builtin_amdgcn_s_wait_asynccnt)
    __builtin_amdgcn_s_wait_asynccnt(0);
#else
    asm volatile("s_wait_asynccnt 0x0" ::: "memory");
#endif
#else
    for (int i = tid; i < 2048; i += 128) dst[i] = src[i];
#endif
  }
  __syncthreads();

  const float scale = 0.03125f;              // 1/sqrt(1024)
  const int jstart = rt & ~1;                // even so the PV kernel's K=32 tiles are covered

  float m8[8], l8[8];
#pragma unroll
  for (int v = 0; v < 8; ++v) { m8[v] = -3.0e38f; l8[v] = 0.0f; }

  // pass 1: online row max / sumexp
  for (int jt = jstart + wave; jt < 256; jt += 4) {
    v8f acc = score_tile(qtile, Kb, jt, half, nidx);
    const int j = jt * 16 + nidx;
#pragma unroll
    for (int v = 0; v < 8; ++v) {
      int i = rt * 16 + v + 8 * half;
      bool keep = (j >= i);
      float sv = keep ? acc[v] * scale : -3.0e38f;
      float mnew = fmaxf(m8[v], redmax16(sv));
      float p = keep ? __expf(sv - mnew) : 0.0f;
      float tsum = redsum16(p);
      l8[v] = l8[v] * __expf(m8[v] - mnew) + tsum;
      m8[v] = mnew;
    }
  }

  // combine wave partials in LDS
  if (nidx == 0) {
#pragma unroll
    for (int v = 0; v < 8; ++v) { mws[wave][half * 8 + v] = m8[v]; lws[wave][half * 8 + v] = l8[v]; }
  }
  __syncthreads();
  float mstar[8], rls[8];
#pragma unroll
  for (int v = 0; v < 8; ++v) {
    int r = half * 8 + v;
    float mm = fmaxf(fmaxf(mws[0][r], mws[1][r]), fmaxf(mws[2][r], mws[3][r]));
    float ll = 0.0f;
#pragma unroll
    for (int w2 = 0; w2 < 4; ++w2) ll += lws[w2][r] * __expf(mws[w2][r] - mm);
    mstar[v] = mm;
    rls[v] = 1.0f / ll;                      // every row keeps at least j == i
  }

  // pass 2: recompute scores, write normalized probs (masked -> 0)
  for (int jt = jstart + wave; jt < 256; jt += 4) {
    v8f acc = score_tile(qtile, Kb, jt, half, nidx);
    const int j = jt * 16 + nidx;
#pragma unroll
    for (int v = 0; v < 8; ++v) {
      int i = rt * 16 + v + 8 * half;
      bool keep = (j >= i);
      float p = keep ? __expf(acc[v] * scale - mstar[v]) * rls[v] : 0.0f;
      Pb[(size_t)i * 4096 + j] = (__bf16)p;
    }
  }
}

// ---------- kernel 3: read = P @ V  -> out[:, 1024:2048] ----------
__global__ void attn_pv_kernel(const __bf16* __restrict__ Pb,
                               const __bf16* __restrict__ Vt,
                               float* __restrict__ out) {
  const int n0 = blockIdx.x * 64;
  const int tid = threadIdx.x;
  const int wave = tid >> 5, lane = tid & 31;
  const int half = lane >> 4, nidx = lane & 15;
  const int r0 = blockIdx.y * 64 + wave * 16;   // this wave's 16 query rows
  const int jlo = (r0 / 32) * 32;               // triangular skip (32-aligned)

  v8f acc[4] = {};
  const char* arow = (const char*)(Pb + (size_t)(r0 + nidx) * 4096);
  const char* vrow0 = (const char*)(Vt + (size_t)(n0 +  0 + nidx) * 4096);
  const char* vrow1 = (const char*)(Vt + (size_t)(n0 + 16 + nidx) * 4096);
  const char* vrow2 = (const char*)(Vt + (size_t)(n0 + 32 + nidx) * 4096);
  const char* vrow3 = (const char*)(Vt + (size_t)(n0 + 48 + nidx) * 4096);
  for (int j0 = jlo; j0 < 4096; j0 += 32) {
    size_t off = (size_t)j0 * 2;
    v16bf a  = load_a(arow + off, half);
    v16bf b0 = load_b(vrow0 + off, half);
    v16bf b1 = load_b(vrow1 + off, half);
    v16bf b2 = load_b(vrow2 + off, half);
    v16bf b3 = load_b(vrow3 + off, half);
    acc[0] = WMMA_BF16(a, b0, acc[0]);
    acc[1] = WMMA_BF16(a, b1, acc[1]);
    acc[2] = WMMA_BF16(a, b2, acc[2]);
    acc[3] = WMMA_BF16(a, b3, acc[3]);
  }

#pragma unroll
  for (int nt = 0; nt < 4; ++nt) {
    int n = n0 + nt * 16 + nidx;
#pragma unroll
    for (int v = 0; v < 8; ++v) {
      int t = r0 + v + 8 * half;
      out[(size_t)t * 2048 + 1024 + n] = acc[nt][v];
    }
  }
}

extern "C" void kernel_launch(void* const* d_in, const int* in_sizes, int n_in,
                              void* d_out, int out_size, void* d_ws, size_t ws_size,
                              hipStream_t stream) {
  const float* x  = (const float*)d_in[0];
  const float* Wk = (const float*)d_in[1];
  const float* bk = (const float*)d_in[2];
  const float* Wq = (const float*)d_in[3];
  const float* bq = (const float*)d_in[4];
  const float* Wv = (const float*)d_in[5];
  const float* bv = (const float*)d_in[6];
  float* out = (float*)d_out;

  char* ws = (char*)d_ws;
  const size_t MB = 1024 * 1024;
  __bf16* Qb = (__bf16*)(ws);                 // 4096x1024 bf16 = 8 MB
  __bf16* Kb = (__bf16*)(ws + 8  * MB);       // 8 MB
  __bf16* Vt = (__bf16*)(ws + 16 * MB);       // 1024x4096 bf16 (transposed) = 8 MB
  __bf16* Pb = (__bf16*)(ws + 24 * MB);       // 4096x4096 bf16 probs = 32 MB

  copy_x_kernel<<<4096, 256, 0, stream>>>(x, out);
  proj_kernel<<<dim3(16, 64, 3), 128, 0, stream>>>(x, Wk, bk, Wq, bq, Wv, bv, Kb, Qb, Vt);
  attn_softmax_kernel<<<256, 128, 0, stream>>>(Qb, Kb, Pb);
  attn_pv_kernel<<<dim3(16, 64), 128, 0, stream>>>(Pb, Vt, out);
}
